// HouseholderFlow_62002147885246
// MI455X (gfx1250) — matrix operand order repaired and checked by reading, not compiled, not run on hardware
//
#include <hip/hip_runtime.h>
#include <hip/hip_bf16.h>

// ---------------------------------------------------------------------------
// v = enc[8192,4096] @ W[1024,4096]^T + b ; out = x - 2 v (v.x)/||v||
// GEMM: compute-bound on MI455X (AI ~312 FLOP/B). bf16 WMMA path:
//   - f32->bf16 RNE on the fly, staged via padded LDS
//   - block tile 256x128, 8 wave32 waves, 64x64 per wave
//     (16 v_wmma_f32_16x16x32_bf16 per wave per K-step, 8 frag loads)
//   - W L2 re-read traffic halved vs 128x128 tiles; 256 WGs of parallelism
// Householder pass: memory-bound streaming kernel, wave32 shfl reduction.
// ---------------------------------------------------------------------------

#define B_ROWS   8192
#define LATENT   1024
#define ENC_DIM  4096

#define BLOCK_M  256
#define BLOCK_N  128
#define KSTEP    32
#define LDS_STRIDE_U 20   // (32 + 8 pad) bf16 per row = 20 uints = 80 bytes

typedef __bf16 bf16_t;
typedef bf16_t v16bf __attribute__((ext_vector_type(16)));
typedef float  v8f   __attribute__((ext_vector_type(8)));

union FragBF {
    v16bf v;
    uint4 q[2];
};

// f32 -> bf16 round-to-nearest-even, packed pair into one dword
__device__ __forceinline__ unsigned int pk_bf16(float a, float b) {
    unsigned int ua = __float_as_uint(a);
    unsigned int ub = __float_as_uint(b);
    ua = (ua + 0x7FFFu + ((ua >> 16) & 1u)) >> 16;
    ub = (ub + 0x7FFFu + ((ub >> 16) & 1u)) >> 16;
    return ua | (ub << 16);
}

__device__ __forceinline__ uint4 pk_bf16x8(const float4& a, const float4& b) {
    return make_uint4(pk_bf16(a.x, a.y), pk_bf16(a.z, a.w),
                      pk_bf16(b.x, b.y), pk_bf16(b.z, b.w));
}

__global__ __launch_bounds__(256, 1)
void gemm_v_kernel(const float* __restrict__ enc,   // [B_ROWS, ENC_DIM]
                   const float* __restrict__ W,     // [LATENT, ENC_DIM]
                   const float* __restrict__ bias,  // [LATENT]
                   float* __restrict__ v)           // [B_ROWS, LATENT]
{
    __shared__ __align__(16) unsigned int As[BLOCK_M * LDS_STRIDE_U]; // 20 KB
    __shared__ __align__(16) unsigned int Bs[BLOCK_N * LDS_STRIDE_U]; // 10 KB

    const int tid  = threadIdx.x;
    const int m0   = blockIdx.y * BLOCK_M;
    const int n0   = blockIdx.x * BLOCK_N;
    const int lane = tid & 31;
    const int wave = tid >> 5;              // 0..7  (4 in M x 2 in N)
    const int wm   = (wave & 3) * 64;       // wave M offset within block
    const int wn   = (wave >> 2) * 64;      // wave N offset within block
    const int fr   = lane & 15;             // fragment row/col within 16
    const bool hi  = lane >= 16;

    // A loader: 1 thread per row, 32 f32 each (rows 0..255)
    const float* gA = enc + (size_t)(m0 + tid) * ENC_DIM;
    unsigned int* sA = &As[tid * LDS_STRIDE_U];
    // B loader: 2 threads per row, 16 f32 each (rows 0..127)
    const int brow = tid >> 1;
    const int bcol = (tid & 1) << 4;        // 0 or 16
    const float* gB = W + (size_t)(n0 + brow) * ENC_DIM + bcol;
    unsigned int* sB = &Bs[brow * LDS_STRIDE_U + (bcol >> 1)];

    v8f acc[4][4] = {};

    for (int k0 = 0; k0 < ENC_DIM; k0 += KSTEP) {
        // ---- global fetch (f32, b128) ----
        float4 a0 = *(const float4*)(gA + k0);
        float4 a1 = *(const float4*)(gA + k0 + 4);
        float4 a2 = *(const float4*)(gA + k0 + 8);
        float4 a3 = *(const float4*)(gA + k0 + 12);
        float4 a4 = *(const float4*)(gA + k0 + 16);
        float4 a5 = *(const float4*)(gA + k0 + 20);
        float4 a6 = *(const float4*)(gA + k0 + 24);
        float4 a7 = *(const float4*)(gA + k0 + 28);
        float4 b0 = *(const float4*)(gB + k0);
        float4 b1 = *(const float4*)(gB + k0 + 4);
        float4 b2 = *(const float4*)(gB + k0 + 8);
        float4 b3 = *(const float4*)(gB + k0 + 12);

        __syncthreads();   // previous iteration's LDS reads done

        // ---- convert to bf16 and stage in LDS ----
        *(uint4*)(sA)      = pk_bf16x8(a0, a1);
        *(uint4*)(sA + 4)  = pk_bf16x8(a2, a3);
        *(uint4*)(sA + 8)  = pk_bf16x8(a4, a5);
        *(uint4*)(sA + 12) = pk_bf16x8(a6, a7);
        *(uint4*)(sB)      = pk_bf16x8(b0, b1);
        *(uint4*)(sB + 4)  = pk_bf16x8(b2, b3);

        __syncthreads();

        // ---- fragments (ds_load_b128) ----
        // A 16x32 bf16: lanes 0-15 row=fr K={0..7,16..23}; lanes 16-31 K={8..15,24..31}
        FragBF afr[4];
#pragma unroll
        for (int i = 0; i < 4; ++i) {
            const unsigned int* p =
                &As[(wm + i * 16 + fr) * LDS_STRIDE_U + (hi ? 4 : 0)];
            afr[i].q[0] = *(const uint4*)(p);
            afr[i].q[1] = *(const uint4*)(p + 8);
        }
        // B 32x16 bf16: lane = column, lanes 0-15 K={0..7,16..23}; lanes 16-31 K={8..15,24..31}
        FragBF bfr[4];
#pragma unroll
        for (int j = 0; j < 4; ++j) {
            const unsigned int* p =
                &Bs[(wn + j * 16 + fr) * LDS_STRIDE_U + (hi ? 4 : 0)];
            bfr[j].q[0] = *(const uint4*)(p);
            bfr[j].q[1] = *(const uint4*)(p + 8);
        }

        // ---- 16 WMMAs per wave per K-step ----
#pragma unroll
        for (int i = 0; i < 4; ++i)
#pragma unroll
            for (int j = 0; j < 4; ++j)
                acc[i][j] = __builtin_amdgcn_wmma_f32_16x16x32_bf16(
                    false, afr[i].v, false, bfr[j].v,
                    (short)0, acc[i][j], false, false);
    }

    // ---- epilogue: bias add, store v ----
    // C/D layout: lanes 0-15 -> N=fr, M=e ; lanes 16-31 -> N=fr, M=e+8
#pragma unroll
    for (int i = 0; i < 4; ++i) {
        const int row0 = m0 + wm + i * 16 + (hi ? 8 : 0);
#pragma unroll
        for (int j = 0; j < 4; ++j) {
            const int col = n0 + wn + j * 16 + fr;
            const float bb = bias[col];
            float* outp = v + (size_t)row0 * LATENT + col;
#pragma unroll
            for (int e = 0; e < 8; ++e)
                outp[(size_t)e * LATENT] = acc[i][j][e] + bb;
        }
    }
}

__global__ __launch_bounds__(256, 1)
void householder_kernel(const float* __restrict__ x,   // [B_ROWS, LATENT]
                        const float* __restrict__ v,   // [B_ROWS, LATENT]
                        float* __restrict__ out,       // [B_ROWS, LATENT]
                        float* __restrict__ logdet)    // [B_ROWS]
{
    const int row = blockIdx.x;
    const int tid = threadIdx.x;          // 256 threads, 4 elems each

    const float4* vp = (const float4*)(v + (size_t)row * LATENT);
    const float4* xp = (const float4*)(x + (size_t)row * LATENT);
    const float4 v4 = vp[tid];
    const float4 x4 = xp[tid];

    float pd = v4.x * x4.x + v4.y * x4.y + v4.z * x4.z + v4.w * x4.w;
    float pn = v4.x * v4.x + v4.y * v4.y + v4.z * v4.z + v4.w * v4.w;

    // wave32 butterfly reduction
#pragma unroll
    for (int off = 16; off > 0; off >>= 1) {
        pd += __shfl_down(pd, off, 32);
        pn += __shfl_down(pn, off, 32);
    }

    __shared__ float sd[8];
    __shared__ float sn[8];
    if ((tid & 31) == 0) {
        sd[tid >> 5] = pd;
        sn[tid >> 5] = pn;
    }
    __syncthreads();

    float dot = 0.f, nrm2 = 0.f;
#pragma unroll
    for (int i = 0; i < 8; ++i) { dot += sd[i]; nrm2 += sn[i]; }

    const float scale = 2.0f * dot * __frsqrt_rn(nrm2);   // 2*dot/||v||

    float4 o;
    o.x = x4.x - scale * v4.x;
    o.y = x4.y - scale * v4.y;
    o.z = x4.z - scale * v4.z;
    o.w = x4.w - scale * v4.w;
    ((float4*)(out + (size_t)row * LATENT))[tid] = o;

    if (tid == 0) logdet[row] = 0.0f;
}

extern "C" void kernel_launch(void* const* d_in, const int* in_sizes, int n_in,
                              void* d_out, int out_size, void* d_ws, size_t ws_size,
                              hipStream_t stream) {
    const float* input = (const float*)d_in[0];   // [8192, 1024]
    const float* enc   = (const float*)d_in[1];   // [8192, 4096]
    const float* W     = (const float*)d_in[2];   // [1024, 4096]
    const float* bias  = (const float*)d_in[3];   // [1024]

    float* out    = (float*)d_out;                         // [8192,1024] ++ [8192,1]
    float* logdet = out + (size_t)B_ROWS * LATENT;
    float* v      = (float*)d_ws;                          // 32 MB scratch

    dim3 g1(LATENT / BLOCK_N, B_ROWS / BLOCK_M);           // (8, 32)
    gemm_v_kernel<<<g1, 256, 0, stream>>>(enc, W, bias, v);

    householder_kernel<<<B_ROWS, 256, 0, stream>>>(input, v, out, logdet);
}
